// MultiHeadedAttention_40269613367835
// MI455X (gfx1250) — matrix operand order repaired and checked by reading, not compiled, not run on hardware
//
#include <hip/hip_runtime.h>
#include <hip/hip_bf16.h>
#include <math.h>

// MultiHeadedAttention forward for MI455X (gfx1250, wave32, WMMA).
// B=2, S=2048, D=1024, H=16, DK=64.
// d_out = [ normed : B*S*D fp32 ][ attn : B*H*S*S fp32 ]
// d_ws  = [ q f16 : B*S*D ][ k f16 : B*S*D ][ v f16 : B*S*D ]  (~25 MB)

#define BB  2
#define SSZ 2048
#define DD  1024
#define HH  16
#define DKK 64
#define BSR (BB * SSZ)   // total rows = 4096

typedef _Float16 v16h __attribute__((ext_vector_type(16)));
typedef _Float16 v8h  __attribute__((ext_vector_type(8)));
typedef float    v8f  __attribute__((ext_vector_type(8)));

union AFrag { v16h v; v8h h[2]; };

__device__ __forceinline__ v8f wmma_f16(v16h a, v16h b, v8f c) {
    // D = A(16x32 f16) * B(32x16 f16) + C(16x16 f32)
    return __builtin_amdgcn_wmma_f32_16x16x32_f16(
        /*neg_a=*/false, a, /*neg_b=*/false, b,
        /*c_mod=*/(short)0, c, /*reuse_a=*/false, /*reuse_b=*/false);
}

// ---------------------------------------------------------------------------
// Kernel 1: Y(f16)[BSR x DD] = X(f32)[BSR x DD] @ W(f32)[DD x DD]
// 256 threads = 8 waves; block tile 64x64; waves arranged 4 (M) x 2 (N, 32 cols).
// ---------------------------------------------------------------------------
__global__ void __launch_bounds__(256)
proj_f16_kernel(const float* __restrict__ X, const float* __restrict__ W,
                _Float16* __restrict__ Y)
{
    __shared__ alignas(32) _Float16 Xs[64][32];   // [m][k]
    __shared__ alignas(32) _Float16 Wt[64][32];   // [n][k] (transposed)

    const int tid  = threadIdx.x;
    const int wid  = tid >> 5;
    const int lane = tid & 31;
    const int lh   = lane >> 4;      // half-wave select
    const int l15  = lane & 15;
    const int m_base = blockIdx.x * 64;
    const int n_base = blockIdx.y * 64;
    const int wm = wid & 3;          // 0..3 -> M sub-tile
    const int wn = wid >> 2;         // 0..1 -> N sub-tile (32 cols)

    // staging coordinates (8 contiguous elements per thread per tile)
    const int xr  = tid >> 2;        // 0..63
    const int xc  = (tid & 3) * 8;   // 0..24
    const int wk  = tid >> 3;        // 0..31
    const int wn0 = (tid & 7) * 8;   // 0..56

    v8f acc0 = {}; v8f acc1 = {};

    for (int k0 = 0; k0 < DD; k0 += 32) {
        {   // stage X tile (fp32 -> f16)
            const float* src = X + (size_t)(m_base + xr) * DD + k0 + xc;
            #pragma unroll
            for (int j = 0; j < 8; ++j) Xs[xr][xc + j] = (_Float16)src[j];
        }
        {   // stage W tile transposed (fp32 -> f16)
            const float* src = W + (size_t)(k0 + wk) * DD + n_base + wn0;
            #pragma unroll
            for (int j = 0; j < 8; ++j) Wt[wn0 + j][wk] = (_Float16)src[j];
        }
        __syncthreads();

        AFrag a;
        a.h[0] = *(const v8h*)&Xs[wm * 16 + l15][lh * 16];
        a.h[1] = *(const v8h*)&Xs[wm * 16 + l15][lh * 16 + 8];
        v16h b0 = *(const v16h*)&Wt[wn * 32 + l15][lh * 16];
        v16h b1 = *(const v16h*)&Wt[wn * 32 + 16 + l15][lh * 16];
        acc0 = wmma_f16(a.v, b0, acc0);
        acc1 = wmma_f16(a.v, b1, acc1);
        __syncthreads();
    }

    #pragma unroll
    for (int g = 0; g < 8; ++g) {
        const int row = m_base + wm * 16 + 8 * lh + g;
        const int c0  = n_base + wn * 32 + l15;
        Y[(size_t)row * DD + c0]      = (_Float16)acc0[g];
        Y[(size_t)row * DD + c0 + 16] = (_Float16)acc1[g];
    }
}

// ---------------------------------------------------------------------------
// Kernel 2: scores = QK^T/8, mask, softmax -> attn fp32.
// grid (S/16, B*H); 256 threads = 8 waves; each wave owns 256 key columns
// (16 WMMA tiles) for the workgroup's 16 query rows. Softmax in registers
// with shfl_xor + small LDS cross-wave reduce.
// ---------------------------------------------------------------------------
__global__ void __launch_bounds__(256)
attn_scores_softmax_kernel(const _Float16* __restrict__ qh,
                           const _Float16* __restrict__ kh,
                           const unsigned char* __restrict__ mask,
                           float* __restrict__ attn)
{
    const int q0  = blockIdx.x * 16;
    const int bh  = blockIdx.y;           // b*H + h
    const int b   = bh / HH;
    const int h   = bh % HH;
    const int tid = threadIdx.x;
    const int wid = tid >> 5;
    const int lane = tid & 31;
    const int lh  = lane >> 4;
    const int l15 = lane & 15;

    __shared__ float red[8][16];          // [wave][row] partials

    // A fragments: Q rows q0..q0+15, DK=64 -> two 16x32 tiles, loaded once.
    AFrag a[2];
    {
        const _Float16* qrow = qh + (size_t)(b * SSZ + q0 + l15) * DD + h * DKK;
        #pragma unroll
        for (int kk = 0; kk < 2; ++kk) {
            a[kk].h[0] = *(const v8h*)(qrow + kk * 32 + lh * 16);
            a[kk].h[1] = *(const v8h*)(qrow + kk * 32 + lh * 16 + 8);
        }
    }

    v8f acc[16];
    #pragma unroll
    for (int t = 0; t < 16; ++t) acc[t] = (v8f){};

    const int nwave = wid * 256;
    #pragma unroll 4
    for (int t = 0; t < 16; ++t) {
        const int n = nwave + t * 16 + l15;     // key column for this lane
        // B fragment: column n of K^T == row n of k-projection; 32 bytes/lane.
        const _Float16* krow = kh + (size_t)(b * SSZ + n) * DD + h * DKK + lh * 16;
        v16h b0 = *(const v16h*)(krow);         // dk tile 0..31
        v16h b1 = *(const v16h*)(krow + 32);    // dk tile 32..63
        acc[t] = wmma_f16(a[0].v, b0, acc[t]);
        acc[t] = wmma_f16(a[1].v, b1, acc[t]);
    }

    // scale + mask
    const float scale = 0.125f;                 // 1/sqrt(64)
    const unsigned char* mb = mask + (size_t)b * SSZ * SSZ;
    #pragma unroll
    for (int t = 0; t < 16; ++t) {
        #pragma unroll
        for (int g = 0; g < 8; ++g) {
            const int row = q0 + 8 * lh + g;
            const int col = nwave + t * 16 + l15;
            float x = acc[t][g] * scale;
            if (mb[(size_t)row * SSZ + col]) x = -__builtin_inff();
            acc[t][g] = x;
        }
    }

    // ---- row max (per lane -> 16-lane group -> cross-wave via LDS) ----
    float rm[8];
    #pragma unroll
    for (int g = 0; g < 8; ++g) {
        float m = acc[0][g];
        #pragma unroll
        for (int t = 1; t < 16; ++t) m = fmaxf(m, acc[t][g]);
        #pragma unroll
        for (int s = 1; s < 16; s <<= 1) m = fmaxf(m, __shfl_xor(m, s, 32));
        rm[g] = m;
    }
    if (l15 == 0) {
        #pragma unroll
        for (int g = 0; g < 8; ++g) red[wid][8 * lh + g] = rm[g];
    }
    __syncthreads();
    float M[8];
    #pragma unroll
    for (int g = 0; g < 8; ++g) {
        float m = -__builtin_inff();
        #pragma unroll
        for (int w = 0; w < 8; ++w) m = fmaxf(m, red[w][8 * lh + g]);
        M[g] = m;
    }
    __syncthreads();

    // ---- exp and row sum ----
    float rs[8];
    #pragma unroll
    for (int g = 0; g < 8; ++g) rs[g] = 0.f;
    #pragma unroll
    for (int t = 0; t < 16; ++t) {
        #pragma unroll
        for (int g = 0; g < 8; ++g) {
            const float p = __expf(acc[t][g] - M[g]);
            acc[t][g] = p;
            rs[g] += p;
        }
    }
    #pragma unroll
    for (int g = 0; g < 8; ++g) {
        #pragma unroll
        for (int s = 1; s < 16; s <<= 1) rs[g] += __shfl_xor(rs[g], s, 32);
    }
    if (l15 == 0) {
        #pragma unroll
        for (int g = 0; g < 8; ++g) red[wid][8 * lh + g] = rs[g];
    }
    __syncthreads();
    float inv[8];
    #pragma unroll
    for (int g = 0; g < 8; ++g) {
        float s = 0.f;
        #pragma unroll
        for (int w = 0; w < 8; ++w) s += red[w][8 * lh + g];
        inv[g] = 1.0f / s;
    }

    // ---- store attn fp32 ----
    float* out = attn + (size_t)bh * SSZ * SSZ;
    #pragma unroll
    for (int t = 0; t < 16; ++t) {
        #pragma unroll
        for (int g = 0; g < 8; ++g) {
            const int row = q0 + 8 * lh + g;
            const int col = nwave + t * 16 + l15;
            out[(size_t)row * SSZ + col] = acc[t][g] * inv[g];
        }
    }
}

// ---------------------------------------------------------------------------
// Kernel 3: res = attn @ V + residual(query).  grid (S/16, B*H); 128 threads
// = 4 waves, each owns one 16x16 tile of the 16x64 head-output block.
// ---------------------------------------------------------------------------
__global__ void __launch_bounds__(128)
attn_v_res_kernel(const float* __restrict__ attn,
                  const _Float16* __restrict__ vh,
                  const float* __restrict__ query,
                  float* __restrict__ res)
{
    const int q0  = blockIdx.x * 16;
    const int bh  = blockIdx.y;
    const int b   = bh / HH;
    const int h   = bh % HH;
    const int tid = threadIdx.x;
    const int wid = tid >> 5;
    const int lane = tid & 31;
    const int lh  = lane >> 4;
    const int l15 = lane & 15;

    __shared__ alignas(32) _Float16 As[16][32];   // attn tile (f16)
    __shared__ alignas(32) _Float16 Vt[64][32];   // V tile transposed [dk][key]

    const float* arow = attn + (size_t)bh * SSZ * SSZ + (size_t)q0 * SSZ;
    v8f acc = {};

    for (int k0 = 0; k0 < SSZ; k0 += 32) {
        {   // stage attn 16x32 fp32 -> f16 (512 elems / 128 thr = 4 each)
            const int e = tid * 4;
            const int r = e >> 5, c = e & 31;
            const float* src = arow + (size_t)r * SSZ + k0 + c;
            #pragma unroll
            for (int j = 0; j < 4; ++j) As[r][c + j] = (_Float16)src[j];
        }
        // stage V (32 keys x 64 dk f16) transposed into Vt
        #pragma unroll
        for (int i = 0; i < 2; ++i) {
            const int chunk = tid + i * 128;      // 0..255
            const int keyr  = chunk >> 3;         // 0..31
            const int dk0   = (chunk & 7) * 8;    // 0..56
            const v8h vv = *(const v8h*)(vh + (size_t)(b * SSZ + k0 + keyr) * DD
                                            + h * DKK + dk0);
            #pragma unroll
            for (int j = 0; j < 8; ++j) Vt[dk0 + j][keyr] = vv[j];
        }
        __syncthreads();

        AFrag a;
        a.h[0] = *(const v8h*)&As[l15][lh * 16];
        a.h[1] = *(const v8h*)&As[l15][lh * 16 + 8];
        v16h bf = *(const v16h*)&Vt[wid * 16 + l15][lh * 16];
        acc = wmma_f16(a.v, bf, acc);
        __syncthreads();
    }

    #pragma unroll
    for (int g = 0; g < 8; ++g) {
        const int row = q0 + 8 * lh + g;
        const int col = h * DKK + wid * 16 + l15;
        const size_t idx = (size_t)(b * SSZ + row) * DD + col;
        res[idx] = acc[g] + query[idx];
    }
}

// ---------------------------------------------------------------------------
// Kernel 4: in-place LayerNorm over D=1024 (biased variance, eps=1e-5).
// ---------------------------------------------------------------------------
__global__ void __launch_bounds__(256)
layernorm_kernel(float* __restrict__ res,
                 const float* __restrict__ gamma,
                 const float* __restrict__ beta)
{
    const int row = blockIdx.x;
    const int tid = threadIdx.x;
    float* r = res + (size_t)row * DD;

    float x[4];
    float s = 0.f, s2 = 0.f;
    #pragma unroll
    for (int j = 0; j < 4; ++j) {
        x[j] = r[tid + j * 256];
        s  += x[j];
        s2 += x[j] * x[j];
    }
    __shared__ float sh1[256];
    __shared__ float sh2[256];
    sh1[tid] = s; sh2[tid] = s2;
    __syncthreads();
    for (int off = 128; off > 0; off >>= 1) {
        if (tid < off) { sh1[tid] += sh1[tid + off]; sh2[tid] += sh2[tid + off]; }
        __syncthreads();
    }
    const float mean = sh1[0] * (1.0f / DD);
    const float var  = sh2[0] * (1.0f / DD) - mean * mean;
    const float rstd = rsqrtf(var + 1e-5f);
    #pragma unroll
    for (int j = 0; j < 4; ++j) {
        const int c = tid + j * 256;
        r[c] = (x[j] - mean) * rstd * gamma[c] + beta[c];
    }
}

// ---------------------------------------------------------------------------
extern "C" void kernel_launch(void* const* d_in, const int* in_sizes, int n_in,
                              void* d_out, int out_size, void* d_ws, size_t ws_size,
                              hipStream_t stream)
{
    const float* key   = (const float*)d_in[0];
    const float* value = (const float*)d_in[1];
    const float* query = (const float*)d_in[2];
    const unsigned char* mask = (const unsigned char*)d_in[3];
    const float* Wq    = (const float*)d_in[4];
    const float* Wk    = (const float*)d_in[5];
    const float* Wv    = (const float*)d_in[6];
    const float* gamma = (const float*)d_in[7];
    const float* beta  = (const float*)d_in[8];

    float* normed = (float*)d_out;                       // [B,S,D]
    float* attn   = normed + (size_t)BSR * DD;           // [B*H,S,S]

    _Float16* qh = (_Float16*)d_ws;                      // [B*S, D] f16
    _Float16* kh = qh + (size_t)BSR * DD;
    _Float16* vh = kh + (size_t)BSR * DD;

    // 1) q/k/v projections (fp32 in -> f16 out, f32-accum WMMA)
    dim3 gproj(BSR / 64, DD / 64);
    proj_f16_kernel<<<gproj, 256, 0, stream>>>(query, Wq, qh);
    proj_f16_kernel<<<gproj, 256, 0, stream>>>(key,   Wk, kh);
    proj_f16_kernel<<<gproj, 256, 0, stream>>>(value, Wv, vh);

    // 2) scores + softmax -> attn (fp32)
    dim3 gatt(SSZ / 16, BB * HH);
    attn_scores_softmax_kernel<<<gatt, 256, 0, stream>>>(qh, kh, mask, attn);

    // 3) attn @ V + residual -> res (in d_out region 1)
    attn_v_res_kernel<<<gatt, 128, 0, stream>>>(attn, vh, query, normed);

    // 4) LayerNorm in place
    layernorm_kernel<<<BSR, 256, 0, stream>>>(normed, gamma, beta);
}
